// MultiHeadAttention_56083682951545
// MI455X (gfx1250) — matrix operand order repaired and checked by reading, not compiled, not run on hardware
//
#include <hip/hip_runtime.h>

// ---------------------------------------------------------------------------
// MultiHeadAttention forward for gfx1250 (MI455X, wave32, WMMA).
// B=4, S=2048, H=512, NH=8, HD=64. fp32 in/out, bf16 WMMA compute (f32 acc).
// Double-buffered LDS pipeline; fp32->bf16 pack via v_fma_mixlo/hi_bf16
// (VOP3P opcodes 62/63, hardware RNE convert into packed register halves).
// ---------------------------------------------------------------------------

#define BM 128
#define BN 128
#define BK 32
#define LDT 40   // LDS row stride in bf16 elems (20 dwords -> conflict-free b128)

typedef __attribute__((ext_vector_type(16))) __bf16 bf16x16;
typedef __attribute__((ext_vector_type(8)))  float  f32x8;

union FragU { bf16x16 v; uint4 q[2]; };

// Pack two f32 into one dword of bf16 (RNE) using the CDNA5 mixed-precision
// FMA converts: v_fma_mixlo_bf16 / v_fma_mixhi_bf16 (D.bf16 = cvt(s0*1.0+0)).
__device__ __forceinline__ unsigned pack_bf16(float lo, float hi) {
  unsigned d;
  asm volatile("v_fma_mixlo_bf16 %0, %1, 1.0, 0\n\t"
               "v_fma_mixhi_bf16 %0, %2, 1.0, 0"
               : "=&v"(d) : "v"(lo), "v"(hi));
  return d;
}

// C[M,N] = alpha * A[M,K] * B[N,K]^T + bias  (NT GEMM, batched over blockIdx.z)
// Per z: off = (z/nh)*perB + (z%nh)*perH for each of A, B, C.
__global__ __launch_bounds__(256)
void gemm_nt_bf16(const float* __restrict__ A,  long long lda, long long aPB, long long aPH,
                  const float* __restrict__ Bm, long long ldb, long long bPB, long long bPH,
                  float*       __restrict__ C,  long long ldc, long long cPB, long long cPH,
                  const float* __restrict__ bias,
                  int M, int N, int K, int nh, float alpha)
{
  __shared__ __align__(16) unsigned short ldsA[2][BM * LDT];
  __shared__ __align__(16) unsigned short ldsB[2][BN * LDT];

  const int z  = blockIdx.z;
  const int zb = z / nh, zh = z % nh;
  A  += (long long)zb * aPB + (long long)zh * aPH;
  Bm += (long long)zb * bPB + (long long)zh * bPH;
  C  += (long long)zb * cPB + (long long)zh * cPH;

  const int m0   = blockIdx.y * BM;
  const int n0   = blockIdx.x * BN;
  const int tid  = threadIdx.x;
  const int lane = tid & 31;
  const int wave = tid >> 5;
  const int wm   = wave & 3;   // 4 waves along M (32 rows each)
  const int wn   = wave >> 2;  // 2 waves along N (64 cols each)

  f32x8 acc[2][4];
  #pragma unroll
  for (int i = 0; i < 2; ++i)
    #pragma unroll
    for (int j = 0; j < 4; ++j)
      #pragma unroll
      for (int r = 0; r < 8; ++r) acc[i][j][r] = 0.0f;

  // Cooperative-load geometry (K-invariant): 1024 float4 per tile, 4 per thread.
  const int lr = tid >> 3;        // base row 0..31 (+ 32*i)
  const int c4 = tid & 7;         // float4 slot within the 32-wide K slab
  bool aOk[4], bOk[4];
  const float* aPtr[4];
  const float* bPtr[4];
  #pragma unroll
  for (int i = 0; i < 4; ++i) {
    int row = lr + i * 32;
    aOk[i] = (m0 + row) < M;
    bOk[i] = (n0 + row) < N;
    aPtr[i] = A  + (long long)(m0 + row) * lda + c4 * 4;
    bPtr[i] = Bm + (long long)(n0 + row) * ldb + c4 * 4;
  }

  // Fragment-load geometry (ISA VGPR layouts):
  const int fr  = lane & 15;
  const int ak0 = (lane >> 4) * 8;    // A frag: K split {k0..k0+7, k0+16..k0+23}
  const int bk0 = (lane >> 4) * 8;    // B frag: K split {k0..k0+7, k0+16..k0+23}

  const int nk = K / BK;
  float4 va[4], vb[4];

  // Prologue: fetch tile 0 and stage into buffer 0.
  #pragma unroll
  for (int i = 0; i < 4; ++i) {
    va[i] = aOk[i] ? *(const float4*)(aPtr[i]) : make_float4(0.f, 0.f, 0.f, 0.f);
    vb[i] = bOk[i] ? *(const float4*)(bPtr[i]) : make_float4(0.f, 0.f, 0.f, 0.f);
  }
  #pragma unroll
  for (int i = 0; i < 4; ++i) {
    uint2 pa; pa.x = pack_bf16(va[i].x, va[i].y); pa.y = pack_bf16(va[i].z, va[i].w);
    uint2 pb; pb.x = pack_bf16(vb[i].x, vb[i].y); pb.y = pack_bf16(vb[i].z, vb[i].w);
    *(uint2*)(&ldsA[0][(lr + i * 32) * LDT + c4 * 4]) = pa;
    *(uint2*)(&ldsB[0][(lr + i * 32) * LDT + c4 * 4]) = pb;
  }
  __syncthreads();

  for (int kt = 0; kt < nk; ++kt) {
    const int cur = kt & 1;
    const int nxt = cur ^ 1;
    const bool more = (kt + 1) < nk;

    // Issue global loads for tile kt+1 first (latency hidden under WMMAs).
    if (more) {
      const int koff = (kt + 1) * BK;
      #pragma unroll
      for (int i = 0; i < 4; ++i) {
        va[i] = aOk[i] ? *(const float4*)(aPtr[i] + koff) : make_float4(0.f, 0.f, 0.f, 0.f);
        vb[i] = bOk[i] ? *(const float4*)(bPtr[i] + koff) : make_float4(0.f, 0.f, 0.f, 0.f);
      }
      if (kt + 2 < nk) {           // speculative L2 prefetch of tile kt+2
        __builtin_prefetch(aPtr[0] + (kt + 2) * BK, 0, 3);
        __builtin_prefetch(bPtr[0] + (kt + 2) * BK, 0, 3);
      }
    }

    // Compute on tile kt from LDS buffer `cur`.
    FragU afrag[2], bfrag[4];
    #pragma unroll
    for (int i = 0; i < 2; ++i) {
      const unsigned short* p = &ldsA[cur][(wm * 32 + i * 16 + fr) * LDT + ak0];
      afrag[i].q[0] = *(const uint4*)(p);
      afrag[i].q[1] = *(const uint4*)(p + 16);
    }
    #pragma unroll
    for (int j = 0; j < 4; ++j) {
      const unsigned short* p = &ldsB[cur][(wn * 64 + j * 16 + fr) * LDT + bk0];
      bfrag[j].q[0] = *(const uint4*)(p);
      bfrag[j].q[1] = *(const uint4*)(p + 16);
    }
    #pragma unroll
    for (int i = 0; i < 2; ++i)
      #pragma unroll
      for (int j = 0; j < 4; ++j)
        acc[i][j] = __builtin_amdgcn_wmma_f32_16x16x32_bf16(
            false, afrag[i].v, false, bfrag[j].v, (short)0, acc[i][j], false, false);

    // Stage tile kt+1 into the other buffer, then one barrier.
    if (more) {
      #pragma unroll
      for (int i = 0; i < 4; ++i) {
        uint2 pa; pa.x = pack_bf16(va[i].x, va[i].y); pa.y = pack_bf16(va[i].z, va[i].w);
        uint2 pb; pb.x = pack_bf16(vb[i].x, vb[i].y); pb.y = pack_bf16(vb[i].z, vb[i].w);
        *(uint2*)(&ldsA[nxt][(lr + i * 32) * LDT + c4 * 4]) = pa;
        *(uint2*)(&ldsB[nxt][(lr + i * 32) * LDT + c4 * 4]) = pb;
      }
      __syncthreads();
    }
  }

  // Epilogue: C/D layout -> row = vgpr + 8*(lane/16), col = lane%16
  const int cr = lane >> 4;
  const int cc = lane & 15;
  #pragma unroll
  for (int j = 0; j < 4; ++j) {
    int nG = n0 + wn * 64 + j * 16 + cc;
    if (nG >= N) continue;
    float bv = bias ? bias[nG] : 0.0f;
    #pragma unroll
    for (int i = 0; i < 2; ++i) {
      int mBase = m0 + wm * 32 + i * 16 + cr * 8;
      #pragma unroll
      for (int r = 0; r < 8; ++r) {
        int mG = mBase + r;
        if (mG < M) C[(long long)mG * ldc + nG] = acc[i][j][r] * alpha + bv;
      }
    }
  }
}

// Vt[b,h,d,s] = Vp[b,s, h*64+d] : 32x32 LDS tile transpose
__global__ __launch_bounds__(256)
void transpose_v(const float* __restrict__ Vp, float* __restrict__ Vt)
{
  __shared__ float tile[32][33];
  const int colBase = blockIdx.x * 32;   // over H = 512
  const int rowBase = blockIdx.y * 32;   // over B*S = 8192
  const int tx = threadIdx.x, ty = threadIdx.y;   // (32, 8)
  #pragma unroll
  for (int i = 0; i < 32; i += 8)
    tile[ty + i][tx] = Vp[(size_t)(rowBase + ty + i) * 512 + colBase + tx];
  __syncthreads();
  #pragma unroll
  for (int i = 0; i < 32; i += 8) {
    int hcol = colBase + ty + i;         // H coordinate
    int rrow = rowBase + tx;             // b*S + s
    int b = rrow >> 11, s = rrow & 2047;
    int h = hcol >> 6,  d = hcol & 63;
    Vt[(((size_t)(b * 8 + h) * 64 + d) << 11) + s] = tile[tx][ty + i];
  }
}

// In-place row softmax over 2048-column rows; one 256-thread block per row.
__global__ __launch_bounds__(256)
void softmax2048(float* __restrict__ att)
{
  float* row = att + (size_t)blockIdx.x * 2048;
  const int tid  = threadIdx.x;
  const int lane = tid & 31;
  const int wid  = tid >> 5;
  __shared__ float smax[8];
  __shared__ float ssum[8];

  float v[8];
  float mx = -3.402823466e38f;
  #pragma unroll
  for (int i = 0; i < 8; ++i) { v[i] = row[tid + i * 256]; mx = fmaxf(mx, v[i]); }
  #pragma unroll
  for (int off = 16; off > 0; off >>= 1) mx = fmaxf(mx, __shfl_xor(mx, off, 32));
  if (lane == 0) smax[wid] = mx;
  __syncthreads();
  mx = smax[0];
  #pragma unroll
  for (int i = 1; i < 8; ++i) mx = fmaxf(mx, smax[i]);

  float s = 0.0f;
  #pragma unroll
  for (int i = 0; i < 8; ++i) { v[i] = __expf(v[i] - mx); s += v[i]; }
  #pragma unroll
  for (int off = 16; off > 0; off >>= 1) s += __shfl_xor(s, off, 32);
  if (lane == 0) ssum[wid] = s;
  __syncthreads();
  s = 0.0f;
  #pragma unroll
  for (int i = 0; i < 8; ++i) s += ssum[i];
  float inv = 1.0f / s;
  #pragma unroll
  for (int i = 0; i < 8; ++i) row[tid + i * 256] = v[i] * inv;
}

extern "C" void kernel_launch(void* const* d_in, const int* in_sizes, int n_in,
                              void* d_out, int out_size, void* d_ws, size_t ws_size,
                              hipStream_t stream)
{
  const int B = 4, S = 2048, H = 512, NH = 8, HD = 64;
  (void)in_sizes; (void)n_in; (void)out_size; (void)ws_size;

  const float* query  = (const float*)d_in[0];
  const float* key_in = (const float*)d_in[1];
  const float* value  = (const float*)d_in[2];
  const float* Wq = (const float*)d_in[3];
  const float* bq = (const float*)d_in[4];
  const float* Wk = (const float*)d_in[5];
  const float* bk = (const float*)d_in[6];
  const float* Wv = (const float*)d_in[7];
  const float* bv = (const float*)d_in[8];
  const float* Wo = (const float*)d_in[9];
  const float* bo = (const float*)d_in[10];

  float* out = (float*)d_out;                              // [B,S,H]
  float* att = out + (size_t)B * S * H;                    // [B,NH,S,S]

  const size_t NE = (size_t)B * S * H;                     // 4,194,304
  float* Qp  = (float*)d_ws;                               // [B,S,H]
  float* Kp  = Qp + NE;                                    // [B,S,H]
  float* Vp  = Kp + NE;                                    // [B,S,H]
  float* Vt  = Vp + NE;                                    // [B,NH,HD,S]
  float* Ctx = Vt + NE;                                    // [B,S,H]

  dim3 blk(256);
  const int MR = B * S;                                    // 8192

  // 1-3. Q/K/V projections: [MR,H] = X[MR,H] * W[H,H]^T + b
  dim3 gProj((H + BN - 1) / BN, (MR + BM - 1) / BM, 1);
  gemm_nt_bf16<<<gProj, blk, 0, stream>>>(query,  H, 0, 0, Wq, H, 0, 0,
                                          Qp, H, 0, 0, bq, MR, H, H, 1, 1.0f);
  gemm_nt_bf16<<<gProj, blk, 0, stream>>>(key_in, H, 0, 0, Wk, H, 0, 0,
                                          Kp, H, 0, 0, bk, MR, H, H, 1, 1.0f);
  gemm_nt_bf16<<<gProj, blk, 0, stream>>>(value,  H, 0, 0, Wv, H, 0, 0,
                                          Vp, H, 0, 0, bv, MR, H, H, 1, 1.0f);

  // 4. Vt[b,h,d,s] = Vp[b,s,h*64+d]
  dim3 gT(H / 32, MR / 32);
  transpose_v<<<gT, dim3(32, 8), 0, stream>>>(Vp, Vt);

  // 5. scores -> att (raw, scaled): per (b,h): S x S = Q[S,64] * K[S,64]^T * 1/8
  dim3 gS((S + BN - 1) / BN, (S + BM - 1) / BM, B * NH);
  gemm_nt_bf16<<<gS, blk, 0, stream>>>(
      Qp, H, (long long)S * H, HD,
      Kp, H, (long long)S * H, HD,
      att, S, (long long)NH * S * S, (long long)S * S,
      nullptr, S, S, HD, NH, 0.125f);

  // 6. softmax in place over att rows
  softmax2048<<<B * NH * S, blk, 0, stream>>>(att);

  // 7. context: per (b,h): [S,64] = att[S,S] * Vt[64,S]^T, written into [B,S,H]
  dim3 gC((HD + BN - 1) / BN, (S + BM - 1) / BM, B * NH);
  gemm_nt_bf16<<<gC, blk, 0, stream>>>(
      att, S, (long long)NH * S * S, (long long)S * S,
      Vt,  S, (long long)NH * HD * S, (long long)HD * S,
      Ctx, H, (long long)S * H, HD,
      nullptr, S, HD, S, NH, 1.0f);

  // 8. output projection: out = Ctx[MR,H] * Wo[H,H]^T + bo
  gemm_nt_bf16<<<gProj, blk, 0, stream>>>(Ctx, H, 0, 0, Wo, H, 0, 0,
                                          out, H, 0, 0, bo, MR, H, H, 1, 1.0f);
}
